// MetaGNN_89842125897828
// MI455X (gfx1250) — matrix operand order repaired and checked by reading, not compiled, not run on hardware
//
#include <hip/hip_runtime.h>

// -------------------------------------------------------------------------
// GraphSAGE on MI455X (gfx1250).
//   * all GEMMs: v_wmma_f32_16x16x32_bf16, one 16x16 C tile per wave32
//   * activations & weights staged as bf16 (halves gather bandwidth of the
//     L2-bound aggregation, removes per-use f32->bf16 conversion VALU)
//   * aggregation: coalesced packed-bf16 gather + fp32 global atomics
//   * mean-divide fused into AGG A-fragment load; bias/relu fused in store
// -------------------------------------------------------------------------

typedef __attribute__((ext_vector_type(16))) __bf16 v16bf;
typedef __attribute__((ext_vector_type(8)))  float  v8f;

#define KDIM 128   // inner dimension of every GEMM in this model

// ---- WMMA fragment builders (CDNA5 ISA 7.12.2 layouts) -------------------
// A 16x32 bf16: lanes 0-15: row M=lane,   K = kblock+{0..7, 16..23}
//               lanes16-31: row M=lane-16, K = kblock+{8..15, 24..31}
__device__ __forceinline__ v16bf load_a_frag_f32(const float* base, int lane,
                                                 int kblock, float scale) {
  const int m  = lane & 15;
  const int kb = (lane < 16) ? 0 : 8;
  const float* p = base + (size_t)m * KDIM + kblock;
  v16bf a;
#pragma unroll
  for (int e = 0; e < 8; ++e) a[e]     = (__bf16)(p[kb + e]      * scale);
#pragma unroll
  for (int e = 0; e < 8; ++e) a[8 + e] = (__bf16)(p[16 + kb + e] * scale);
  return a;
}

__device__ __forceinline__ v16bf load_a_frag_bf(const __bf16* base, int lane,
                                                int kblock) {
  const int m  = lane & 15;
  const int kb = (lane < 16) ? 0 : 8;
  const __bf16* p = base + (size_t)m * KDIM + kblock;
  v16bf a;
#pragma unroll
  for (int e = 0; e < 8; ++e) a[e]     = p[kb + e];
#pragma unroll
  for (int e = 0; e < 8; ++e) a[8 + e] = p[16 + kb + e];
  return a;
}

// B 32x16 bf16 with B[k][n] = W[n][k] (W row-major [N,K], stored bf16).
// lanes 0-15: col N=n0+lane, K = kblock+{0..7, 16..23} ; lanes16-31: K = kblock+{8..15, 24..31}
__device__ __forceinline__ v16bf load_b_frag_bf(const __bf16* __restrict__ W,
                                                int lane, int n0, int kblock) {
  const int n  = n0 + (lane & 15);
  const int kb = (lane < 16) ? 0 : 8;
  const __bf16* p = W + (size_t)n * KDIM + kblock + kb;
  v16bf b;
#pragma unroll
  for (int e = 0; e < 8; ++e) { b[e] = p[e]; b[8 + e] = p[16 + e]; }
  return b;
}

// ---- stage 1: h0_bf16 = relu(x_f32 @ W_bf16^T + bias) --------------------
__global__ void k_gemm1(const float* __restrict__ A, const __bf16* __restrict__ W,
                        const float* __restrict__ bias, __bf16* __restrict__ out) {
  const int lane = threadIdx.x & 31;
  const int wave = threadIdx.x >> 5;
  const int m0 = blockIdx.x * 16;
  const int n0 = wave * 16;
  const float* Ab = A + (size_t)m0 * KDIM;
  v8f acc = {};
#pragma unroll
  for (int kb = 0; kb < KDIM; kb += 32) {
    v16bf a = load_a_frag_f32(Ab, lane, kb, 1.0f);
    v16bf b = load_b_frag_bf(W, lane, n0, kb);
    acc = __builtin_amdgcn_wmma_f32_16x16x32_bf16(false, a, false, b,
                                                  (short)0, acc, false, false);
  }
  const int n  = n0 + (lane & 15);
  const int mb = m0 + ((lane < 16) ? 0 : 8);
  const float bn = bias[n];
#pragma unroll
  for (int v = 0; v < 8; ++v)
    out[(size_t)(mb + v) * KDIM + n] = (__bf16)fmaxf(acc[v] + bn, 0.0f);
}

// ---- SAGE layer: out = act( (AGG/deg) @ Wl^T + bl + H @ Wr^T ) -----------
__global__ void k_sage(const float* __restrict__ AGG, const float* __restrict__ deg,
                       const __bf16* __restrict__ H, const __bf16* __restrict__ Wl,
                       const float* __restrict__ bl, const __bf16* __restrict__ Wr,
                       void* __restrict__ out, int N, int do_relu, int out_bf16) {
  const int lane = threadIdx.x & 31;
  const int wave = threadIdx.x >> 5;
  const int m0 = blockIdx.x * 16;
  const int n0 = wave * 16;
  const int m  = m0 + (lane & 15);
  const float dscale = 1.0f / fmaxf(deg[m], 1.0f);
  const float*  Ab = AGG + (size_t)m0 * KDIM;
  const __bf16* Hb = H   + (size_t)m0 * KDIM;
  __builtin_prefetch(Hb + (size_t)(lane & 15) * KDIM, 0, 1);  // global_prefetch_b8
  v8f acc = {};
#pragma unroll
  for (int kb = 0; kb < KDIM; kb += 32) {
    v16bf a = load_a_frag_f32(Ab, lane, kb, dscale);
    v16bf b = load_b_frag_bf(Wl, lane, n0, kb);
    acc = __builtin_amdgcn_wmma_f32_16x16x32_bf16(false, a, false, b,
                                                  (short)0, acc, false, false);
  }
#pragma unroll
  for (int kb = 0; kb < KDIM; kb += 32) {
    v16bf a = load_a_frag_bf(Hb, lane, kb);
    v16bf b = load_b_frag_bf(Wr, lane, n0, kb);
    acc = __builtin_amdgcn_wmma_f32_16x16x32_bf16(false, a, false, b,
                                                  (short)0, acc, false, false);
  }
  const int n  = n0 + (lane & 15);
  const int mb = m0 + ((lane < 16) ? 0 : 8);
  const float bn = bl[n];
#pragma unroll
  for (int v = 0; v < 8; ++v) {
    float r = acc[v] + bn;
    if (do_relu) r = fmaxf(r, 0.0f);
    const size_t idx = (size_t)(mb + v) * N + n;
    if (out_bf16) ((__bf16*)out)[idx] = (__bf16)r;
    else          ((float*)out)[idx]  = r;
  }
}

// ---- helpers -------------------------------------------------------------
__global__ void k_zero_f32(float* __restrict__ p, size_t n) {
  size_t i = (size_t)blockIdx.x * blockDim.x + threadIdx.x;
  size_t stride = (size_t)gridDim.x * blockDim.x;
  for (; i < n; i += stride) p[i] = 0.0f;
}

__global__ void k_cvt_bf16(const float* __restrict__ s, __bf16* __restrict__ d, int n) {
  int i = blockIdx.x * blockDim.x + threadIdx.x;
  if (i < n) d[i] = (__bf16)s[i];
}

__global__ void k_deg(const long long* __restrict__ dst, float* __restrict__ deg, int E) {
  int e = blockIdx.x * blockDim.x + threadIdx.x;
  if (e < E) atomicAdd(&deg[dst[e]], 1.0f);
}

// One wave32 per edge row. Each lane gathers one packed-bf16 dword (128B
// coalesced per wave per chunk), unpacks (bf16 bits << 16 == f32 bits) and
// does two fp32 atomics into the L2-resident fp32 accumulation buffer.
__global__ void k_scatter_add_bf(const long long* __restrict__ src,
                                 const long long* __restrict__ dst,
                                 const __bf16* __restrict__ feat,
                                 float* __restrict__ out, int E) {
  const int lane = threadIdx.x & 31;
  const int wave = (blockIdx.x * blockDim.x + threadIdx.x) >> 5;
  const int EPW = 8;
  int e = wave * EPW;
#pragma unroll 1
  for (int i = 0; i < EPW; ++i, ++e) {
    if (e >= E) return;
    const unsigned int* fs =
        (const unsigned int*)(feat + (size_t)src[e] * KDIM);   // 64 dwords/row
    float* fo = out + (size_t)dst[e] * KDIM;
#pragma unroll
    for (int c = 0; c < 2; ++c) {
      const int d = lane + 32 * c;
      const unsigned int u = fs[d];
      atomicAdd(&fo[2 * d],     __uint_as_float(u << 16));
      atomicAdd(&fo[2 * d + 1], __uint_as_float(u & 0xffff0000u));
    }
  }
}

// Row-wise log_softmax over 64 columns, in place: one wave per row.
__global__ void k_log_softmax64(float* __restrict__ out, int M) {
  const int lane = threadIdx.x & 31;
  const int wave = threadIdx.x >> 5;
  const int row  = blockIdx.x * (blockDim.x >> 5) + wave;
  if (row >= M) return;
  float* p = out + (size_t)row * 64;
  float v0 = p[lane], v1 = p[lane + 32];
  float mx = fmaxf(v0, v1);
#pragma unroll
  for (int off = 16; off > 0; off >>= 1) mx = fmaxf(mx, __shfl_xor(mx, off, 32));
  float s = __expf(v0 - mx) + __expf(v1 - mx);
#pragma unroll
  for (int off = 16; off > 0; off >>= 1) s += __shfl_xor(s, off, 32);
  const float lse = mx + __logf(s);
  p[lane]      = v0 - lse;
  p[lane + 32] = v1 - lse;
}

// -------------------------------------------------------------------------
extern "C" void kernel_launch(void* const* d_in, const int* in_sizes, int n_in,
                              void* d_out, int out_size, void* d_ws, size_t ws_size,
                              hipStream_t stream) {
  (void)n_in; (void)out_size; (void)ws_size;
  const float*     x     = (const float*)d_in[0];
  const long long* ei    = (const long long*)d_in[1];   // int64 [2, E]
  const float*     lin_W = (const float*)d_in[2];
  const float*     lin_b = (const float*)d_in[3];
  const float*     c1_Wl = (const float*)d_in[4];
  const float*     c1_bl = (const float*)d_in[5];
  const float*     c1_Wr = (const float*)d_in[6];
  const float*     c2_Wl = (const float*)d_in[7];
  const float*     c2_bl = (const float*)d_in[8];
  const float*     c2_Wr = (const float*)d_in[9];

  const int M = in_sizes[0] / KDIM;   // 100000
  const int E = in_sizes[1] / 2;      // 1600000
  const long long* src = ei;
  const long long* dst = ei + E;
  const size_t feats = (size_t)M * KDIM;

  // workspace: [agg f32 | h0 bf16 | h1 bf16 | deg f32 | 5x bf16 weights]
  float*  aggF = (float*)d_ws;
  __bf16* h0b  = (__bf16*)(aggF + feats);
  __bf16* h1b  = h0b + feats;
  float*  deg  = (float*)(h1b + feats);
  __bf16* wbuf = (__bf16*)(deg + M);
  __bf16* lin_Wb = wbuf;                  // 128*128
  __bf16* c1_Wlb = lin_Wb + 128 * 128;
  __bf16* c1_Wrb = c1_Wlb + 128 * 128;
  __bf16* c2_Wlb = c1_Wrb + 128 * 128;    // 64*128
  __bf16* c2_Wrb = c2_Wlb + 64 * 128;

  const int mtiles = M / 16;              // 6250 (M % 16 == 0)
  float* logits = (float*)d_out;

  // degrees (shared by both SAGE layers)
  k_zero_f32<<<256, 256, 0, stream>>>(deg, (size_t)M);
  k_deg<<<(E + 255) / 256, 256, 0, stream>>>(dst, deg, E);

  // one-time weight conversion to bf16 (131 KB total)
  k_cvt_bf16<<<64, 256, 0, stream>>>(lin_W, lin_Wb, 128 * 128);
  k_cvt_bf16<<<64, 256, 0, stream>>>(c1_Wl, c1_Wlb, 128 * 128);
  k_cvt_bf16<<<64, 256, 0, stream>>>(c1_Wr, c1_Wrb, 128 * 128);
  k_cvt_bf16<<<32, 256, 0, stream>>>(c2_Wl, c2_Wlb, 64 * 128);
  k_cvt_bf16<<<32, 256, 0, stream>>>(c2_Wr, c2_Wrb, 64 * 128);

  // stage 1: h0 = relu(x @ lin_W^T + lin_b)  -> bf16
  k_gemm1<<<mtiles, 256, 0, stream>>>(x, lin_Wb, lin_b, h0b);

  // stage 2: SAGEConv1 -> h1 (bf16)
  k_zero_f32<<<8192, 256, 0, stream>>>(aggF, feats);
  {
    int waves  = (E + 7) / 8;
    int blocks = (waves * 32 + 255) / 256;
    k_scatter_add_bf<<<blocks, 256, 0, stream>>>(src, dst, h0b, aggF, E);
  }
  k_sage<<<mtiles, 256, 0, stream>>>(aggF, deg, h0b, c1_Wlb, c1_bl, c1_Wrb,
                                     h1b, KDIM, 1, 1);

  // stage 3: SAGEConv2 -> logits f32 in d_out (agg buffer reused)
  k_zero_f32<<<8192, 256, 0, stream>>>(aggF, feats);
  {
    int waves  = (E + 7) / 8;
    int blocks = (waves * 32 + 255) / 256;
    k_scatter_add_bf<<<blocks, 256, 0, stream>>>(src, dst, h1b, aggF, E);
  }
  k_sage<<<mtiles, 128, 0, stream>>>(aggF, deg, h1b, c2_Wlb, c2_bl, c2_Wrb,
                                     logits, 64, 0, 0);

  // stage 4: log_softmax rows (in place on d_out)
  k_log_softmax64<<<(M + 7) / 8, 256, 0, stream>>>(logits, M);
}
